// LiteTracker_13572096656116
// MI455X (gfx1250) — compile-verified, matrix operand order checked by reading, not run on hardware
//
#include <hip/hip_runtime.h>
#include <math.h>

typedef __attribute__((ext_vector_type(16))) _Float16 v16h;
typedef __attribute__((ext_vector_type(8)))  _Float16 v8h;
typedef __attribute__((ext_vector_type(8)))  float    v8f;

#define GRID     7
#define GG       49
#define CH       128
#define NTRK     4096
#define K1       2401
#define KPAD     2432      // 76 * 32
#define H1       384
#define H2       256
#define TRK_WG   16
#define NWAVE    4
#define TRK_WAVE 4

// ---------------------------------------------------------------------------
// WMMA fragment loader (16-bit A or B operand) from a row-major f16 matrix.
// ISA 7.12.2 layout: lanes 0-15 hold row (lane&15), K = 0..7 then 16..23;
// lanes 16-31 hold K = 8..15 then 24..31.  Two contiguous 16B loads per lane.
// ---------------------------------------------------------------------------
__device__ __forceinline__ v16h load_frag(const _Float16* p, int ld, int lane) {
    const int r  = lane & 15;
    const int kb = (lane & 16) >> 1;          // 0 or 8
    const _Float16* q = p + r * ld + kb;
    v16h out;
    *((v8h*)&out)       = *((const v8h*)q);        // K = kb .. kb+7
    *(((v8h*)&out) + 1) = *((const v8h*)(q + 16)); // K = kb+16 .. kb+23
    return out;
}

__device__ __forceinline__ v8f wmma_f16(v16h a, v16h b, v8f c) {
    return __builtin_amdgcn_wmma_f32_16x16x32_f16(false, a, false, b,
                                                  (short)0, c, false, false);
}

// ---------------------------------------------------------------------------
// Prep: transpose + f32->f16 convert the MLP weights once.
//   w1t[n][k] (n<384, k<KPAD, zero-padded K) ; w2t[n][k] (n<256, k<384)
// ---------------------------------------------------------------------------
__global__ void lt_prep(const float* __restrict__ w1, const float* __restrict__ w2,
                        _Float16* __restrict__ w1t, _Float16* __restrict__ w2t) {
    const int stride = gridDim.x * blockDim.x;
    const int idx    = blockIdx.x * blockDim.x + threadIdx.x;
    for (int i = idx; i < H1 * KPAD; i += stride) {
        int n = i / KPAD, k = i - n * KPAD;
        float v = (k < K1) ? w1[(size_t)k * H1 + n] : 0.0f;
        w1t[i] = (_Float16)v;
    }
    for (int i = idx; i < H2 * H1; i += stride) {
        int n = i / H1, k = i - n * H1;
        w2t[i] = (_Float16)w2[(size_t)k * H2 + n];
    }
}

// ---------------------------------------------------------------------------
// Fused per-level kernel: bilinear gather -> 49x49 correlation (WMMA)
//                         -> MLP1+GELU (WMMA) -> MLP2 (WMMA) -> out
// ---------------------------------------------------------------------------
__global__ __launch_bounds__(128)
void lt_level(const float* __restrict__ fmap,    // [CH, Hm, Wm]
              const float* __restrict__ tfeat,   // [49, NTRK, CH]
              const float* __restrict__ coords,  // [NTRK, 2]
              const _Float16* __restrict__ w1t,  // [H1, KPAD]
              const float* __restrict__ b1,
              const _Float16* __restrict__ w2t,  // [H2, H1]
              const float* __restrict__ b2,
              float* __restrict__ out,           // [NTRK, 1024]
              int Hm, int Wm, int lvl) {
    extern __shared__ _Float16 smem[];
    const int tid  = threadIdx.x;
    const int wave = tid >> 5;
    const int lane = tid & 31;

    _Float16* cf = smem + (size_t)wave * 64 * CH;                 // [64][128] per wave
    _Float16* tm = smem + (size_t)NWAVE * 64 * CH + (size_t)wave * 64 * CH;
    _Float16* X  = smem + (size_t)2 * NWAVE * 64 * CH;            // [16][KPAD]
    _Float16* Hs = X + (size_t)TRK_WG * KPAD;                     // [16][384]

    // Zero pad rows 49..63 of per-wave scratch; zero X cols 2401..2431.
    for (int i = lane; i < 15 * CH; i += 32) {
        cf[GG * CH + i] = (_Float16)0.0f;
        tm[GG * CH + i] = (_Float16)0.0f;
    }
    for (int i = tid; i < TRK_WG * (KPAD - K1); i += 128) {
        int r = i / (KPAD - K1), c = i - r * (KPAD - K1);
        X[r * KPAD + K1 + c] = (_Float16)0.0f;
    }

    const int   n0    = blockIdx.x * TRK_WG;
    const float scale = 1.0f / (float)(1 << lvl);
    const int   HW    = Hm * Wm;

    // ---------------- Stage 1: gather + per-track correlation ----------------
    for (int it = 0; it < TRK_WAVE; ++it) {
        const int tl = wave * TRK_WAVE + it;      // local row in X tile
        const int n  = n0 + tl;
        const float cx = coords[2 * n + 0] * scale;
        const float cy = coords[2 * n + 1] * scale;

        // Template support features -> f16 LDS (coalesced stream, read once)
        for (int p = 0; p < GG; ++p) {
            const float* src = tfeat + ((size_t)p * NTRK + n) * CH;
#pragma unroll
            for (int q = 0; q < 4; ++q) {
                int c = lane + 32 * q;
                tm[p * CH + c] = (_Float16)src[c];
            }
        }
        // Bilinear-sampled correlation features -> f16 LDS
        for (int p = 0; p < GG; ++p) {
            const int di = p / GRID - 3;          // added to x (varies along i)
            const int dj = p % GRID - 3;          // added to y (varies along j)
            const float xs = cx + (float)di;
            const float ys = cy + (float)dj;
            const float x0f = floorf(xs), y0f = floorf(ys);
            const float wx = xs - x0f, wy = ys - y0f;
            const int x0 = (int)x0f, y0 = (int)y0f;
            const int x1 = x0 + 1,  y1 = y0 + 1;
            const bool vx0 = (x0 >= 0) && (x0 < Wm);
            const bool vx1 = (x1 >= 0) && (x1 < Wm);
            const bool vy0 = (y0 >= 0) && (y0 < Hm);
            const bool vy1 = (y1 >= 0) && (y1 < Hm);
            const int xc0 = min(max(x0, 0), Wm - 1);
            const int xc1 = min(max(x1, 0), Wm - 1);
            const int yc0 = min(max(y0, 0), Hm - 1);
            const int yc1 = min(max(y1, 0), Hm - 1);
            const float w00 = (vy0 && vx0) ? (1.0f - wy) * (1.0f - wx) : 0.0f;
            const float w01 = (vy0 && vx1) ? (1.0f - wy) * wx          : 0.0f;
            const float w10 = (vy1 && vx0) ? wy * (1.0f - wx)          : 0.0f;
            const float w11 = (vy1 && vx1) ? wy * wx                   : 0.0f;
#pragma unroll
            for (int q = 0; q < 4; ++q) {
                int c = lane + 32 * q;
                const float* fb = fmap + (size_t)c * HW;
                float v = w00 * fb[yc0 * Wm + xc0] + w01 * fb[yc0 * Wm + xc1]
                        + w10 * fb[yc1 * Wm + xc0] + w11 * fb[yc1 * Wm + xc1];
                cf[p * CH + c] = (_Float16)v;
            }
        }

        // Correlation: corr_vol[hw][ij] = cf(64x128,row-major) . tm^T
        const int mloc = (lane & 16) >> 1;        // C/D row offset (0 or 8)
        const int nloc = lane & 15;
        for (int mt = 0; mt < 4; ++mt) {
            for (int nt = 0; nt < 4; ++nt) {
                v8f acc = {};
#pragma unroll
                for (int ks = 0; ks < 4; ++ks) {
                    v16h a = load_frag(cf + mt * 16 * CH + ks * 32, CH, lane);
                    v16h b = load_frag(tm + nt * 16 * CH + ks * 32, CH, lane);
                    acc = wmma_f16(a, b, acc);
                }
#pragma unroll
                for (int v = 0; v < 8; ++v) {
                    int hw = mt * 16 + mloc + v;
                    int ij = nt * 16 + nloc;
                    if (hw < GG && ij < GG)
                        X[tl * KPAD + hw * GG + ij] = (_Float16)acc[v];
                }
            }
        }
    }

    __syncthreads();

    // ---------------- Stage 2: MLP1 (16x2432 @ 2432x384) + GELU --------------
    {
        v8f zero = {};
        v8f acc[6];
#pragma unroll
        for (int t = 0; t < 6; ++t) acc[t] = zero;
        for (int ks = 0; ks < KPAD / 32; ++ks) {
            v16h a = load_frag(X + ks * 32, KPAD, lane);   // shared across N-tiles
#pragma unroll
            for (int t = 0; t < 6; ++t) {
                int nt = wave + 4 * t;                     // 24 N-tiles over 4 waves
                v16h b = load_frag(w1t + (size_t)nt * 16 * KPAD + ks * 32, KPAD, lane);
                acc[t] = wmma_f16(a, b, acc[t]);
            }
        }
        const int mloc = (lane & 16) >> 1;
        const int nloc = lane & 15;
#pragma unroll
        for (int t = 0; t < 6; ++t) {
            int nt = wave + 4 * t;
#pragma unroll
            for (int v = 0; v < 8; ++v) {
                int m    = mloc + v;
                int ncol = nt * 16 + nloc;
                float hv = acc[t][v] + b1[ncol];
                hv = 0.5f * hv * (1.0f + erff(hv * 0.70710678118654752f)); // exact GELU
                Hs[m * H1 + ncol] = (_Float16)hv;
            }
        }
    }

    __syncthreads();

    // ---------------- Stage 3: MLP2 (16x384 @ 384x256) -> global -------------
    {
        v8f zero = {};
        v8f acc[4];
#pragma unroll
        for (int t = 0; t < 4; ++t) acc[t] = zero;
        for (int ks = 0; ks < H1 / 32; ++ks) {
            v16h a = load_frag(Hs + ks * 32, H1, lane);
#pragma unroll
            for (int t = 0; t < 4; ++t) {
                int nt = wave + 4 * t;                     // 16 N-tiles over 4 waves
                v16h b = load_frag(w2t + (size_t)nt * 16 * H1 + ks * 32, H1, lane);
                acc[t] = wmma_f16(a, b, acc[t]);
            }
        }
        const int mloc = (lane & 16) >> 1;
        const int nloc = lane & 15;
#pragma unroll
        for (int t = 0; t < 4; ++t) {
            int nt = wave + 4 * t;
#pragma unroll
            for (int v = 0; v < 8; ++v) {
                int m    = mloc + v;
                int ncol = nt * 16 + nloc;
                out[(size_t)(n0 + m) * 1024 + (size_t)lvl * 256 + ncol] =
                    acc[t][v] + b2[ncol];
            }
        }
    }
}

// ---------------------------------------------------------------------------
// Launch
// ---------------------------------------------------------------------------
extern "C" void kernel_launch(void* const* d_in, const int* in_sizes, int n_in,
                              void* d_out, int out_size, void* d_ws, size_t ws_size,
                              hipStream_t stream) {
    (void)in_sizes; (void)n_in; (void)out_size; (void)ws_size;
    // setup_inputs() dict order: fmaps0,tfeat0, fmaps1,tfeat1, ..., coords, w1,b1,w2,b2
    const float* fmaps[4]  = {(const float*)d_in[0], (const float*)d_in[2],
                              (const float*)d_in[4], (const float*)d_in[6]};
    const float* tfeats[4] = {(const float*)d_in[1], (const float*)d_in[3],
                              (const float*)d_in[5], (const float*)d_in[7]};
    const float* coords = (const float*)d_in[8];
    const float* w1 = (const float*)d_in[9];
    const float* b1 = (const float*)d_in[10];
    const float* w2 = (const float*)d_in[11];
    const float* b2 = (const float*)d_in[12];
    float* out = (float*)d_out;

    _Float16* w1t = (_Float16*)d_ws;                      // [384][2432] f16
    _Float16* w2t = w1t + (size_t)H1 * KPAD;              // [256][384]  f16

    lt_prep<<<256, 256, 0, stream>>>(w1, w2, w1t, w2t);

    static const int Hms[4] = {96, 48, 24, 12};
    static const int Wms[4] = {128, 64, 32, 16};
    const size_t smem_bytes =
        (size_t)(2 * NWAVE * 64 * CH + TRK_WG * KPAD + TRK_WG * H1) * sizeof(_Float16);
    for (int lvl = 0; lvl < 4; ++lvl) {
        lt_level<<<NTRK / TRK_WG, 128, smem_bytes, stream>>>(
            fmaps[lvl], tfeats[lvl], coords, w1t, b1, w2t, b2, out,
            Hms[lvl], Wms[lvl], lvl);
    }
}
